// Ithemal_12730283065402
// MI455X (gfx1250) — compile-verified, matrix operand order checked by reading, not run on hardware
//
#include <hip/hip_runtime.h>
#include <math.h>

// ---------------- model dims ----------------
#define L_LAYERS 6
#define D_MODEL  768
#define H_HEADS  12
#define F_FF     3072
#define N_INSTR  256
#define T_TOK    32
#define HD_DIM   64

// gfx1250 async global->LDS copies (ASYNCcnt-tracked). Verified to assemble
// in round 2 (ds/vmem counts dropped accordingly). Set to 0 for sync fallback.
#ifndef USE_ASYNC_LDS
#define USE_ASYNC_LDS 1
#endif

// ---------------- CDNA5 vector types ----------------
typedef __bf16 bf16_t;
typedef __attribute__((ext_vector_type(16))) __bf16        v16bf;
typedef __attribute__((ext_vector_type(8)))  float         v8f;
typedef __attribute__((ext_vector_type(8)))  unsigned int  v8u;
typedef __attribute__((ext_vector_type(4)))  unsigned int  v4u;

// fp32 <-> bf16 helpers (bit manipulation, RNE)
__device__ __forceinline__ bf16_t f2bf(float f) {
  unsigned int u = __builtin_bit_cast(unsigned int, f);
  unsigned int r = (u + 0x7FFFu + ((u >> 16) & 1u)) >> 16;
  return __builtin_bit_cast(bf16_t, (unsigned short)r);
}
__device__ __forceinline__ float bf2f(bf16_t b) {
  unsigned int u = ((unsigned int)__builtin_bit_cast(unsigned short, b)) << 16;
  return __builtin_bit_cast(float, u);
}

// 16-byte global -> LDS copy.
// GLOBAL_LOAD_ASYNC_TO_LDS_B128: VDST = per-lane LDS byte address (low 32 bits
// of the flat address per the aperture rule), VADDR = 64-bit global address.
__device__ __forceinline__ void cp16_g2l(bf16_t* lds_dst, const bf16_t* gsrc) {
#if USE_ASYNC_LDS
  unsigned int laddr = (unsigned int)(uintptr_t)lds_dst;
  asm volatile("global_load_async_to_lds_b128 %0, %1, off"
               :: "v"(laddr), "v"(gsrc) : "memory");
#else
  *(v4u*)lds_dst = *(const v4u*)gsrc;
#endif
}
__device__ __forceinline__ void cp_fence() {
#if USE_ASYNC_LDS
  asm volatile("s_wait_asynccnt 0x0" ::: "memory");
#endif
}

// ---------------- weight conversion + transpose ----------------
// in: [K,N] fp32  ->  out: [N,K] bf16 (GEMM B-tiles contiguous along K)
__global__ __launch_bounds__(256) void cvt_w_transpose(const float* __restrict__ in,
                                                       bf16_t* __restrict__ out,
                                                       int K, int N) {
  int total = K * N;
  for (int idx = blockIdx.x * 256 + threadIdx.x; idx < total; idx += gridDim.x * 256) {
    int k = idx / N, n = idx - k * N;
    out[(size_t)n * K + k] = f2bf(in[idx]);
  }
}

// ---------------- embedding ----------------
__global__ __launch_bounds__(256) void embed_kernel(const int* __restrict__ ids,
                                                    const float* __restrict__ emb,
                                                    const float* __restrict__ pos,
                                                    bf16_t* __restrict__ X) {
  int idx = blockIdx.x * 256 + threadIdx.x;   // over N*T*D
  int d  = idx % D_MODEL;
  int nt = idx / D_MODEL;
  int t  = nt % T_TOK;
  X[idx] = f2bf(emb[(size_t)ids[nt] * D_MODEL + d] + pos[t * D_MODEL + d]);
}

// ---------------- last-token gather + instr_pos ----------------
__global__ __launch_bounds__(256) void gather_instr(const bf16_t* __restrict__ X,
                                                    const float* __restrict__ ipos,
                                                    bf16_t* __restrict__ Xi) {
  int idx = blockIdx.x * 256 + threadIdx.x;   // over N*D
  int d = idx % D_MODEL;
  int n = idx / D_MODEL;
  Xi[idx] = f2bf(bf2f(X[(size_t)(n * T_TOK + T_TOK - 1) * D_MODEL + d]) + ipos[idx]);
}

// ---------------- WMMA GEMM (bf16 activations, double-buffered async LDS) ----------------
// X: [M,K] bf16, WT: [N,K] bf16 (pre-transposed), out/res: [M,N] bf16, bias fp32
#define BM 128
#define BN 128
#define BK 32
#define LDH 40          // bf16 per LDS row (32 + 8 pad); 80B row, 16B aligned

__device__ __forceinline__ void stage_tiles(bf16_t* As, bf16_t* Bs,
                                            const bf16_t* __restrict__ X,
                                            const bf16_t* __restrict__ WT,
                                            int m0, int n0, int k0, int K, int tid) {
  // 128 rows x 64B per tile -> 512 x 16B chunks, 2 per thread per tile
  #pragma unroll
  for (int j = 0; j < 2; ++j) {
    int c = j * 256 + tid;
    int r = c >> 2, seg = c & 3;
    cp16_g2l(As + r * LDH + seg * 8, X + (size_t)(m0 + r) * K + k0 + seg * 8);
  }
  #pragma unroll
  for (int j = 0; j < 2; ++j) {
    int c = j * 256 + tid;
    int r = c >> 2, seg = c & 3;
    cp16_g2l(Bs + r * LDH + seg * 8, WT + (size_t)(n0 + r) * K + k0 + seg * 8);
  }
}

template <bool GELU, bool RES>
__global__ __launch_bounds__(256) void gemm_wmma_bf16(
    const bf16_t* __restrict__ X, const bf16_t* __restrict__ WT,
    const float* __restrict__ bias, const bf16_t* __restrict__ res,
    bf16_t* __restrict__ out, int K, int Nn)
{
  __shared__ bf16_t As[2][BM * LDH];   // ping-pong A tiles [m][k]
  __shared__ bf16_t Bs[2][BN * LDH];   // ping-pong B tiles [n][k]

  const int tid  = threadIdx.x;
  const int lane = tid & 31;
  const int wave = tid >> 5;
  const int wm   = wave & 3;        // 4 waves along M (32 rows each)
  const int wn   = wave >> 2;       // 2 waves along N (64 cols each)
  const int m0   = blockIdx.y * BM;
  const int n0   = blockIdx.x * BN;
  const int half = lane >> 4;
  const int l16  = lane & 15;

  const v8f vzero = {};
  v8f acc[2][4];
  #pragma unroll
  for (int i = 0; i < 2; ++i)
    #pragma unroll
    for (int j = 0; j < 4; ++j) acc[i][j] = vzero;

  // prologue: stage tile 0
  stage_tiles(As[0], Bs[0], X, WT, m0, n0, 0, K, tid);
  cp_fence();
  __syncthreads();

  int cur = 0;
  for (int k0 = 0; k0 < K; k0 += BK) {
    // fragment loads (ISA 16-bit A layout: lane row = l16,
    // dword i holds K-pair {2i,2i+1}+8*half for i<4, +16 for i>=4)
    v16bf af[2], bfr[4];
    #pragma unroll
    for (int fm = 0; fm < 2; ++fm) {
      const unsigned int* src =
          (const unsigned int*)(As[cur]) + (size_t)(wm * 32 + fm * 16 + l16) * (LDH / 2);
      v8u u;
      #pragma unroll
      for (int i = 0; i < 8; ++i) {
        int p = ((i < 4) ? i : 8 + (i - 4)) + 4 * half;
        u[i] = src[p];
      }
      af[fm] = __builtin_bit_cast(v16bf, u);
    }
    #pragma unroll
    for (int fn = 0; fn < 4; ++fn) {
      const unsigned int* src =
          (const unsigned int*)(Bs[cur]) + (size_t)(wn * 64 + fn * 16 + l16) * (LDH / 2);
      v8u u;
      #pragma unroll
      for (int i = 0; i < 8; ++i) {
        int p = ((i < 4) ? i : 8 + (i - 4)) + 4 * half;
        u[i] = src[p];
      }
      bfr[fn] = __builtin_bit_cast(v16bf, u);
    }

    // kick off async DMA of the NEXT tile while this tile computes
    if (k0 + BK < K)
      stage_tiles(As[cur ^ 1], Bs[cur ^ 1], X, WT, m0, n0, k0 + BK, K, tid);

    #pragma unroll
    for (int fm = 0; fm < 2; ++fm)
      #pragma unroll
      for (int fn = 0; fn < 4; ++fn)
        acc[fm][fn] = __builtin_amdgcn_wmma_f32_16x16x32_bf16(
            false, af[fm], false, bfr[fn], (short)0, acc[fm][fn], false, false);

    cp_fence();       // this wave's async copies have landed in LDS
    __syncthreads();  // all waves done reading cur & done staging nxt
    cur ^= 1;
  }

  // epilogue: C/D layout — lane column n = l16, dword i row m = i + 8*half
  #pragma unroll
  for (int fm = 0; fm < 2; ++fm) {
    #pragma unroll
    for (int fn = 0; fn < 4; ++fn) {
      int n = n0 + wn * 64 + fn * 16 + l16;
      float bv = bias[n];
      #pragma unroll
      for (int i = 0; i < 8; ++i) {
        int m = m0 + wm * 32 + fm * 16 + half * 8 + i;
        float v = acc[fm][fn][i] + bv;
        if (GELU) v = 0.5f * v * (1.0f + erff(v * 0.70710678118654752f));
        if (RES)  v += bf2f(res[(size_t)m * Nn + n]);
        out[(size_t)m * Nn + n] = f2bf(v);
      }
    }
  }
}

// ---------------- causal attention (bf16 I/O, fp32 math; negligible FLOPs) ----------------
__global__ __launch_bounds__(256) void attn_kernel(
    const bf16_t* __restrict__ Q, const bf16_t* __restrict__ Km,
    const bf16_t* __restrict__ V, bf16_t* __restrict__ O, int B, int S)
{
  __shared__ float qrow[8][HD_DIM];
  __shared__ float probs[8][256];
  const int bh = blockIdx.x;
  const int b = bh / H_HEADS, h = bh % H_HEADS;
  const int lane = threadIdx.x & 31, wave = threadIdx.x >> 5;
  const size_t base = (size_t)b * S * D_MODEL + h * HD_DIM;
  const int nIter = S / 8;   // S in {32, 256}: uniform across waves

  for (int it = 0; it < nIter; ++it) {
    const int sq = it * 8 + wave;
    const bf16_t* qp = Q + base + (size_t)sq * D_MODEL;
    qrow[wave][lane]      = bf2f(qp[lane]);
    qrow[wave][lane + 32] = bf2f(qp[lane + 32]);
    __syncthreads();

    for (int j = lane; j <= sq; j += 32) {
      const bf16_t* kp = Km + base + (size_t)j * D_MODEL;
      float s = 0.f;
      #pragma unroll 8
      for (int d = 0; d < HD_DIM; ++d) s += qrow[wave][d] * bf2f(kp[d]);
      probs[wave][j] = s * 0.125f;    // 1/sqrt(64)
    }
    __syncthreads();

    if (lane == 0) {
      float mx = -3.0e38f;
      for (int j = 0; j <= sq; ++j) mx = fmaxf(mx, probs[wave][j]);
      float sum = 0.f;
      for (int j = 0; j <= sq; ++j) { float e = __expf(probs[wave][j] - mx); probs[wave][j] = e; sum += e; }
      float inv = 1.f / sum;
      for (int j = 0; j <= sq; ++j) probs[wave][j] *= inv;
    }
    __syncthreads();

    bf16_t* op = O + base + (size_t)sq * D_MODEL;
    for (int d = lane; d < HD_DIM; d += 32) {
      float s = 0.f;
      for (int j = 0; j <= sq; ++j) s += probs[wave][j] * bf2f(V[base + (size_t)j * D_MODEL + d]);
      op[d] = f2bf(s);
    }
    __syncthreads();
  }
}

// ---------------- final scalar prediction ----------------
__global__ __launch_bounds__(256) void pred_kernel(const bf16_t* __restrict__ X,
                                                   const float* __restrict__ W,
                                                   const float* __restrict__ b,
                                                   float* __restrict__ out) {
  __shared__ float red[256];
  int t = threadIdx.x;
  float s = 0.f;
  for (int d = t; d < D_MODEL; d += 256)
    s += bf2f(X[(size_t)(N_INSTR - 1) * D_MODEL + d]) * W[d];
  red[t] = s;
  __syncthreads();
  for (int off = 128; off > 0; off >>= 1) {
    if (t < off) red[t] += red[t + off];
    __syncthreads();
  }
  if (t == 0) out[0] = red[0] + b[0];
}

// ---------------- host orchestration ----------------
struct StageW {
  const float *Wqkv, *bqkv, *Wo, *bo, *Wp, *bp, *W1, *b1, *W2, *b2;
};

enum { EP_PLAIN = 0, EP_GELU = 1, EP_RES = 2 };

static void launch_gemm(const bf16_t* Xp, const float* Wf, const float* bias,
                        const bf16_t* res, bf16_t* out, int M, int K, int Nn,
                        int mode, bf16_t* wbfT, hipStream_t s) {
  cvt_w_transpose<<<2048, 256, 0, s>>>(Wf, wbfT, K, Nn);
  dim3 grid(Nn / BN, M / BM);
  if (mode == EP_GELU)
    gemm_wmma_bf16<true, false><<<grid, 256, 0, s>>>(Xp, wbfT, bias, res, out, K, Nn);
  else if (mode == EP_RES)
    gemm_wmma_bf16<false, true><<<grid, 256, 0, s>>>(Xp, wbfT, bias, res, out, K, Nn);
  else
    gemm_wmma_bf16<false, false><<<grid, 256, 0, s>>>(Xp, wbfT, bias, res, out, K, Nn);
}

static void run_stage(bf16_t* X, bf16_t* Qb, bf16_t* Kb, bf16_t* Vb, bf16_t* AOb,
                      bf16_t* FFb, bf16_t* wbfT, const StageW& w,
                      int Bseq, int Sseq, hipStream_t s) {
  const int M  = Bseq * Sseq;
  const size_t DD = (size_t)D_MODEL * D_MODEL;
  const size_t DF = (size_t)D_MODEL * F_FF;
  for (int l = 0; l < L_LAYERS; ++l) {
    const float* Wq = w.Wqkv + (size_t)l * 3 * DD;
    const float* bq = w.bqkv + (size_t)l * 3 * D_MODEL;
    // QKV projections
    launch_gemm(X, Wq,          bq,               nullptr, Qb, M, D_MODEL, D_MODEL, EP_PLAIN, wbfT, s);
    launch_gemm(X, Wq + DD,     bq + D_MODEL,     nullptr, Kb, M, D_MODEL, D_MODEL, EP_PLAIN, wbfT, s);
    launch_gemm(X, Wq + 2 * DD, bq + 2 * D_MODEL, nullptr, Vb, M, D_MODEL, D_MODEL, EP_PLAIN, wbfT, s);
    // causal attention
    attn_kernel<<<Bseq * H_HEADS, 256, 0, s>>>(Qb, Kb, Vb, AOb, Bseq, Sseq);
    // output proj then attn-proj with residual into X (in-place safe)
    launch_gemm(AOb, w.Wo + (size_t)l * DD, w.bo + l * D_MODEL, nullptr, Qb, M, D_MODEL, D_MODEL, EP_PLAIN, wbfT, s);
    launch_gemm(Qb,  w.Wp + (size_t)l * DD, w.bp + l * D_MODEL, X,       X,  M, D_MODEL, D_MODEL, EP_RES,   wbfT, s);
    // FFN: gelu(X@W1+b1) @ W2 + b2, residual into X
    launch_gemm(X,   w.W1 + (size_t)l * DF, w.b1 + (size_t)l * F_FF, nullptr, FFb, M, D_MODEL, F_FF, EP_GELU, wbfT, s);
    launch_gemm(FFb, w.W2 + (size_t)l * DF, w.b2 + l * D_MODEL,      X,       X,   M, F_FF, D_MODEL, EP_RES,  wbfT, s);
  }
}

extern "C" void kernel_launch(void* const* d_in, const int* in_sizes, int n_in,
                              void* d_out, int out_size, void* d_ws, size_t ws_size,
                              hipStream_t stream) {
  (void)in_sizes; (void)n_in; (void)out_size; (void)ws_size;
  const int*   token_ids = (const int*)d_in[0];
  const float* tok_embed = (const float*)d_in[1];
  const float* tok_pos   = (const float*)d_in[2];
  const float* instr_pos = (const float*)d_in[3];
  StageW tw = { (const float*)d_in[4],  (const float*)d_in[5],  (const float*)d_in[6],
                (const float*)d_in[7],  (const float*)d_in[8],  (const float*)d_in[9],
                (const float*)d_in[10], (const float*)d_in[11], (const float*)d_in[12],
                (const float*)d_in[13] };
  StageW iw = { (const float*)d_in[14], (const float*)d_in[15], (const float*)d_in[16],
                (const float*)d_in[17], (const float*)d_in[18], (const float*)d_in[19],
                (const float*)d_in[20], (const float*)d_in[21], (const float*)d_in[22],
                (const float*)d_in[23] };
  const float* pred_W = (const float*)d_in[24];
  const float* pred_b = (const float*)d_in[25];

  // workspace layout (bf16 activations)
  char* wsb = (char*)d_ws;
  size_t off = 0;
  auto take = [&](size_t bytes) -> char* {
    char* p = wsb + off;
    off = (off + bytes + 255) & ~(size_t)255;
    return p;
  };
  const size_t MT = (size_t)N_INSTR * T_TOK;            // 8192 token rows
  bf16_t* X    = (bf16_t*)take(MT * D_MODEL * 2);
  bf16_t* Qb   = (bf16_t*)take(MT * D_MODEL * 2);
  bf16_t* Kb   = (bf16_t*)take(MT * D_MODEL * 2);
  bf16_t* Vb   = (bf16_t*)take(MT * D_MODEL * 2);
  bf16_t* AOb  = (bf16_t*)take(MT * D_MODEL * 2);
  bf16_t* FFb  = (bf16_t*)take(MT * F_FF * 2);
  bf16_t* wbfT = (bf16_t*)take((size_t)D_MODEL * F_FF * 2);  // largest weight (transposed)
  bf16_t* X2   = (bf16_t*)take((size_t)N_INSTR * D_MODEL * 2);

  // token stage: [256,32,768]
  embed_kernel<<<(MT * D_MODEL) / 256, 256, 0, stream>>>(token_ids, tok_embed, tok_pos, X);
  run_stage(X, Qb, Kb, Vb, AOb, FFb, wbfT, tw, N_INSTR, T_TOK, stream);

  // instruction stage: [1,256,768]
  gather_instr<<<((size_t)N_INSTR * D_MODEL) / 256, 256, 0, stream>>>(X, instr_pos, X2);
  run_stage(X2, Qb, Kb, Vb, AOb, FFb, wbfT, iw, 1, N_INSTR, stream);

  // scalar prediction from last row
  pred_kernel<<<1, 256, 0, stream>>>(X2, pred_W, pred_b, (float*)d_out);
}
